// StressDecoder_15075335209404
// MI455X (gfx1250) — matrix operand order repaired and checked by reading, not compiled, not run on hardware
//
#include <hip/hip_runtime.h>

// ---------------- constants ----------------
constexpr int cB = 1024, cS = 50, cH = 256, cT = 512, cD = 512, cDFF = 2048, cNH = 4;

typedef __bf16 bf16;
typedef __attribute__((ext_vector_type(16))) __bf16 v16bf;
typedef __attribute__((ext_vector_type(8)))  __bf16 v8bf;
typedef __attribute__((ext_vector_type(8)))  float  v8f;
typedef __attribute__((ext_vector_type(2)))  float  v2f;

__device__ inline v16bf mk16(v8bf lo, v8bf hi) {
  v16bf r;
#pragma unroll
  for (int e = 0; e < 8; ++e) { r[e] = lo[e]; r[8 + e] = hi[e]; }
  return r;
}

// ---------------- generic bf16 WMMA GEMM: C[M,N] = A[M,K] @ W[N,K]^T + bias ----------------
// grid = (M/16, N/64), block = 32 (one wave). Optional relu, f32 and/or bf16 outputs.
__global__ void __launch_bounds__(32)
k_gemm_bf16(const bf16* __restrict__ A, const bf16* __restrict__ W,
            const float* __restrict__ bias, float* __restrict__ Cf,
            bf16* __restrict__ Cbf, int N, int K, int relu)
{
  const int m0 = blockIdx.x << 4;
  const int n0 = blockIdx.y << 6;
  const int lane = threadIdx.x;
  const int h = lane >> 4;
  const int r = lane & 15;
  const bf16* arow = A + (size_t)(m0 + r) * K;
  const bf16* w0 = W + (size_t)(n0 + r) * K;
  const bf16* w1 = w0 + (size_t)16 * K;
  const bf16* w2 = w0 + (size_t)32 * K;
  const bf16* w3 = w0 + (size_t)48 * K;
  v8f c0 = {}, c1 = {}, c2 = {}, c3 = {};
  if (bias) {
    float b0 = bias[n0 + r], b1 = bias[n0 + 16 + r];
    float b2 = bias[n0 + 32 + r], b3 = bias[n0 + 48 + r];
#pragma unroll
    for (int j = 0; j < 8; ++j) { c0[j] = b0; c1[j] = b1; c2[j] = b2; c3[j] = b3; }
  }
  for (int k0 = 0; k0 < K; k0 += 32) {
    // A fragment: contiguous 16 at k0+16h of A row m (matches B)
    v16bf a = mk16(*(const v8bf*)(arow + k0 + 16 * h),
                   *(const v8bf*)(arow + k0 + 16 * h + 8));
    // B fragment (same k order as A): contiguous 16 at k0+16h of W row n
    const int bo = k0 + 16 * h;
    v16bf b0 = mk16(*(const v8bf*)(w0 + bo), *(const v8bf*)(w0 + bo + 8));
    v16bf b1 = mk16(*(const v8bf*)(w1 + bo), *(const v8bf*)(w1 + bo + 8));
    v16bf b2 = mk16(*(const v8bf*)(w2 + bo), *(const v8bf*)(w2 + bo + 8));
    v16bf b3 = mk16(*(const v8bf*)(w3 + bo), *(const v8bf*)(w3 + bo + 8));
    c0 = __builtin_amdgcn_wmma_f32_16x16x32_bf16(false, a, false, b0, (short)0, c0, false, false);
    c1 = __builtin_amdgcn_wmma_f32_16x16x32_bf16(false, a, false, b1, (short)0, c1, false, false);
    c2 = __builtin_amdgcn_wmma_f32_16x16x32_bf16(false, a, false, b2, (short)0, c2, false, false);
    c3 = __builtin_amdgcn_wmma_f32_16x16x32_bf16(false, a, false, b3, (short)0, c3, false, false);
  }
#pragma unroll
  for (int j = 0; j < 8; ++j) {
    int row = m0 + j + 8 * h;               // C layout: VGPR j -> M = j + 8*half, N = lane%16
    size_t base = (size_t)row * N + n0 + r;
    float v0 = c0[j], v1 = c1[j], v2 = c2[j], v3 = c3[j];
    if (relu) { v0 = fmaxf(v0, 0.f); v1 = fmaxf(v1, 0.f); v2 = fmaxf(v2, 0.f); v3 = fmaxf(v3, 0.f); }
    if (Cf)  { Cf[base] = v0; Cf[base + 16] = v1; Cf[base + 32] = v2; Cf[base + 48] = v3; }
    if (Cbf) { Cbf[base] = (bf16)v0; Cbf[base + 16] = (bf16)v1;
               Cbf[base + 32] = (bf16)v2; Cbf[base + 48] = (bf16)v3; }
  }
}

// ---------------- f32 WMMA GEMM (16x16x4) for the GRU recurrence ----------------
__global__ void __launch_bounds__(32)
k_gemm_f32(const float* __restrict__ A, const float* __restrict__ W,
           const float* __restrict__ bias, float* __restrict__ Cf, int N, int K)
{
  const int m0 = blockIdx.x << 4;
  const int n0 = blockIdx.y << 6;
  const int lane = threadIdx.x;
  const int h = lane >> 4;
  const int r = lane & 15;
  const float* arow = A + (size_t)(m0 + r) * K;
  const float* w0 = W + (size_t)(n0 + r) * K;
  const float* w1 = w0 + (size_t)16 * K;
  const float* w2 = w0 + (size_t)32 * K;
  const float* w3 = w0 + (size_t)48 * K;
  v8f c0 = {}, c1 = {}, c2 = {}, c3 = {};
  if (bias) {
    float b0 = bias[n0 + r], b1 = bias[n0 + 16 + r];
    float b2 = bias[n0 + 32 + r], b3 = bias[n0 + 48 + r];
#pragma unroll
    for (int j = 0; j < 8; ++j) { c0[j] = b0; c1[j] = b1; c2[j] = b2; c3[j] = b3; }
  }
  for (int k0 = 0; k0 < K; k0 += 4) {
    const int ao = k0 + 2 * h;              // f32 A/B layout: pair at k0 + 2*half
    v2f a  = *(const v2f*)(arow + ao);
    v2f b0 = *(const v2f*)(w0 + ao);
    v2f b1 = *(const v2f*)(w1 + ao);
    v2f b2 = *(const v2f*)(w2 + ao);
    v2f b3 = *(const v2f*)(w3 + ao);
    c0 = __builtin_amdgcn_wmma_f32_16x16x4_f32(false, a, false, b0, (short)0, c0, false, false);
    c1 = __builtin_amdgcn_wmma_f32_16x16x4_f32(false, a, false, b1, (short)0, c1, false, false);
    c2 = __builtin_amdgcn_wmma_f32_16x16x4_f32(false, a, false, b2, (short)0, c2, false, false);
    c3 = __builtin_amdgcn_wmma_f32_16x16x4_f32(false, a, false, b3, (short)0, c3, false, false);
  }
#pragma unroll
  for (int j = 0; j < 8; ++j) {
    int row = m0 + j + 8 * h;
    size_t base = (size_t)row * N + n0 + r;
    Cf[base] = c0[j]; Cf[base + 16] = c1[j]; Cf[base + 32] = c2[j]; Cf[base + 48] = c3[j];
  }
}

// ---------------- small utility kernels ----------------
__global__ void k_cvt(const float* __restrict__ s, bf16* __restrict__ d, size_t n) {
  size_t i = (size_t)blockIdx.x * 256 + threadIdx.x;
  if (i < n) d[i] = (bf16)s[i];
}

__global__ void k_zero(float* __restrict__ p, size_t n) {
  size_t i = (size_t)blockIdx.x * 256 + threadIdx.x;
  if (i < n) p[i] = 0.f;
}

__global__ void k_addpos(const float* __restrict__ syl, const float* __restrict__ pos,
                         bf16* __restrict__ x) {
  int idx = blockIdx.x * 256 + threadIdx.x;    // < B*S*H
  int i = idx & (cH - 1);
  int s = (idx >> 8) % cS;
  x[idx] = (bf16)(syl[idx] + pos[s * cH + i]);
}

// GRU pointwise step (PyTorch gate order r,z,n). idx over B*H.
__global__ void k_gru_step(const float* __restrict__ xg, const float* __restrict__ hg,
                           float* __restrict__ h, float* __restrict__ outf,
                           bf16* __restrict__ outbf, int s, int dirOff)
{
  int idx = blockIdx.x * 256 + threadIdx.x;
  int b = idx >> 8;
  int i = idx & (cH - 1);
  size_t xrow = ((size_t)b * cS + s) * (3 * cH);
  size_t hrow = (size_t)b * (3 * cH);
  float xr = xg[xrow + i], xz = xg[xrow + cH + i], xn = xg[xrow + 2 * cH + i];
  float hr = hg[hrow + i], hz = hg[hrow + cH + i], hn = hg[hrow + 2 * cH + i];
  float rg = 1.f / (1.f + __expf(-(xr + hr)));
  float zg = 1.f / (1.f + __expf(-(xz + hz)));
  float ng = tanhf(xn + rg * hn);
  float hnew = (1.f - zg) * ng + zg * h[idx];
  h[idx] = hnew;
  size_t ob = ((size_t)b * cS + s) * (2 * cH) + dirOff + i;
  outf[ob] = hnew;
  outbf[ob] = (bf16)hnew;
}

// LayerNorm over last dim C (256 or 512), one block (256 thr) per row.
// out = LN(x (+res)) * g + b ; optional relu ; optional postadd (added after relu).
__global__ void __launch_bounds__(256)
k_ln(const float* __restrict__ x, const float* __restrict__ res,
     const float* __restrict__ g, const float* __restrict__ bta,
     float* __restrict__ outf, bf16* __restrict__ outbf,
     const float* __restrict__ postadd, int relu, int C)
{
  __shared__ float sbuf[256];
  const int row = blockIdx.x;
  const int tid = threadIdx.x;
  const int nper = C >> 8;                    // 1 or 2
  const float* xr = x + (size_t)row * C;
  const float* rr = res ? res + (size_t)row * C : nullptr;
  float vals[2];
  float s = 0.f;
  for (int p = 0; p < nper; ++p) {
    int i = tid + (p << 8);
    float v = xr[i] + (rr ? rr[i] : 0.f);
    vals[p] = v; s += v;
  }
  sbuf[tid] = s; __syncthreads();
  for (int o = 128; o > 0; o >>= 1) { if (tid < o) sbuf[tid] += sbuf[tid + o]; __syncthreads(); }
  float mean = sbuf[0] / C;
  __syncthreads();
  float s2 = 0.f;
  for (int p = 0; p < nper; ++p) { float d = vals[p] - mean; s2 += d * d; }
  sbuf[tid] = s2; __syncthreads();
  for (int o = 128; o > 0; o >>= 1) { if (tid < o) sbuf[tid] += sbuf[tid + o]; __syncthreads(); }
  float rstd = rsqrtf(sbuf[0] / C + 1e-5f);
  for (int p = 0; p < nper; ++p) {
    int i = tid + (p << 8);
    float o = (vals[p] - mean) * rstd * g[i] + bta[i];
    if (relu) o = fmaxf(o, 0.f);
    if (postadd) o += postadd[(size_t)row * C + i];
    if (outf)  outf[(size_t)row * C + i] = o;
    if (outbf) outbf[(size_t)row * C + i] = (bf16)o;
  }
}

// Self-attention (S=50, dh=128, NH=4): one block per (b, head).
__global__ void __launch_bounds__(256)
k_selfattn(const bf16* __restrict__ qkv, bf16* __restrict__ ctx)
{
  __shared__ bf16 tq[cS * 128];
  __shared__ bf16 tk[cS * 128];
  __shared__ float pr[cS * cS];
  const int b = blockIdx.x >> 2;
  const int head = blockIdx.x & 3;
  const int tid = threadIdx.x;
  const size_t rs = 3 * cD;                   // 1536
  const size_t base = (size_t)b * cS * rs + head * 128;
  for (int idx = tid; idx < cS * 128; idx += 256) {
    int j = idx >> 7, d = idx & 127;
    tq[idx] = qkv[base + (size_t)j * rs + d];
    tk[idx] = qkv[base + (size_t)j * rs + cD + d];
  }
  __syncthreads();
  const float scale = 0.08838834764831845f;   // 1/sqrt(128)
  for (int idx = tid; idx < cS * cS; idx += 256) {
    int i = idx / cS, j = idx % cS;
    float acc = 0.f;
    for (int d = 0; d < 128; ++d) acc += (float)tq[i * 128 + d] * (float)tk[j * 128 + d];
    pr[idx] = acc * scale;
  }
  __syncthreads();
  if (tid < cS) {
    float mx = -3.4e38f;
    for (int j = 0; j < cS; ++j) mx = fmaxf(mx, pr[tid * cS + j]);
    float sum = 0.f;
    for (int j = 0; j < cS; ++j) { float e = __expf(pr[tid * cS + j] - mx); pr[tid * cS + j] = e; sum += e; }
    float inv = 1.f / sum;
    for (int j = 0; j < cS; ++j) pr[tid * cS + j] *= inv;
  }
  __syncthreads();
  for (int idx = tid; idx < cS * 128; idx += 256) {        // reuse tq for V
    int j = idx >> 7, d = idx & 127;
    tq[idx] = qkv[base + (size_t)j * rs + 2 * cD + d];
  }
  __syncthreads();
  for (int idx = tid; idx < cS * 128; idx += 256) {
    int i = idx >> 7, d = idx & 127;
    float acc = 0.f;
    for (int j = 0; j < cS; ++j) acc += pr[i * cS + j] * (float)tq[j * 128 + d];
    ctx[((size_t)b * cS + i) * cD + head * 128 + d] = (bf16)acc;
  }
}

// Cross-attention (T=512, dh=64): one block (64 thr) per (b, head, query i).
__global__ void __launch_bounds__(64)
k_crossattn(const float* __restrict__ q2, const bf16* __restrict__ kv, bf16* __restrict__ ctx)
{
  __shared__ float qv[64];
  __shared__ float sc[cT];
  __shared__ float red[64];
  int blk = blockIdx.x;
  int i = blk % cS; int tmp = blk / cS;
  int head = tmp & 3; int b = tmp >> 2;
  int tid = threadIdx.x;
  qv[tid] = q2[((size_t)b * cS + i) * cH + head * 64 + tid];
  __syncthreads();
  const size_t kvbase = (size_t)b * cT * (2 * cH) + head * 64;
  for (int j = tid; j < cT; j += 64) {
    const bf16* kr = kv + kvbase + (size_t)j * (2 * cH);
    float acc = 0.f;
    for (int d = 0; d < 64; ++d) acc += qv[d] * (float)kr[d];
    sc[j] = acc * 0.125f;                    // 1/sqrt(64)
  }
  __syncthreads();
  float lm = -3.4e38f;
  for (int j = tid; j < cT; j += 64) lm = fmaxf(lm, sc[j]);
  red[tid] = lm; __syncthreads();
  for (int o = 32; o > 0; o >>= 1) { if (tid < o) red[tid] = fmaxf(red[tid], red[tid + o]); __syncthreads(); }
  float mx = red[0]; __syncthreads();
  float ls = 0.f;
  for (int j = tid; j < cT; j += 64) { float e = __expf(sc[j] - mx); sc[j] = e; ls += e; }
  red[tid] = ls; __syncthreads();
  for (int o = 32; o > 0; o >>= 1) { if (tid < o) red[tid] += red[tid + o]; __syncthreads(); }
  float inv = 1.f / red[0]; __syncthreads();
  for (int j = tid; j < cT; j += 64) sc[j] *= inv;
  __syncthreads();
  float acc = 0.f;
  for (int j = 0; j < cT; ++j)
    acc += sc[j] * (float)kv[kvbase + (size_t)j * (2 * cH) + cH + tid];  // V at col 256+head*64+d
  ctx[((size_t)b * cS + i) * cH + head * 64 + tid] = (bf16)acc;
}

// enhanced = sc + ctxout ; logits = enhanced @ cls_w^T + cls_b  (one thread per row)
__global__ void k_final(const float* __restrict__ sc, const float* __restrict__ ctxo,
                        const float* __restrict__ cw, const float* __restrict__ cb,
                        float* __restrict__ out)
{
  int r = blockIdx.x * 256 + threadIdx.x;   // < B*S
  float a0 = cb[0], a1 = cb[1];
  const float* s = sc + (size_t)r * cH;
  const float* c = ctxo + (size_t)r * cH;
  for (int i = 0; i < cH; ++i) {
    float e = s[i] + c[i];
    a0 += e * cw[i];
    a1 += e * cw[cH + i];
  }
  out[(size_t)r * 2]     = a0;
  out[(size_t)r * 2 + 1] = a1;
}

// ---------------- host-side launch helpers ----------------
static void gemm_bf(hipStream_t st, const bf16* A, const bf16* W, const float* bias,
                    float* Cf, bf16* Cbf, int M, int N, int K, int relu) {
  dim3 g(M / 16, N / 64);
  k_gemm_bf16<<<g, dim3(32), 0, st>>>(A, W, bias, Cf, Cbf, N, K, relu);
}
static void gemm_f32(hipStream_t st, const float* A, const float* W, const float* bias,
                     float* Cf, int M, int N, int K) {
  dim3 g(M / 16, N / 64);
  k_gemm_f32<<<g, dim3(32), 0, st>>>(A, W, bias, Cf, N, K);
}

extern "C" void kernel_launch(void* const* d_in, const int* in_sizes, int n_in,
                              void* d_out, int out_size, void* d_ws, size_t ws_size,
                              hipStream_t stream) {
  (void)in_sizes; (void)n_in; (void)out_size; (void)ws_size;
  const float* syl   = (const float*)d_in[0];
  const float* phon  = (const float*)d_in[1];
  const float* enc   = (const float*)d_in[2];
  const float* pose  = (const float*)d_in[3];

  char* w = (char*)d_ws;
  size_t off = 0;
  auto alloc = [&](size_t bytes) -> void* {
    void* p = w + off;
    off += (bytes + 255) & ~(size_t)255;
    return p;
  };

  // bf16 weight copies
  bf16* wihf_bf   = (bf16*)alloc((size_t)768 * 256 * 2);
  bf16* wihb_bf   = (bf16*)alloc((size_t)768 * 256 * 2);
  bf16* tfinw_bf  = (bf16*)alloc((size_t)1536 * 512 * 2);
  bf16* tfoutw_bf = (bf16*)alloc((size_t)512 * 512 * 2);
  bf16* lin1w_bf  = (bf16*)alloc((size_t)2048 * 512 * 2);
  bf16* lin2w_bf  = (bf16*)alloc((size_t)512 * 2048 * 2);
  bf16* scw_bf    = (bf16*)alloc((size_t)256 * 512 * 2);
  bf16* ppw_bf    = (bf16*)alloc((size_t)256 * 256 * 2);
  bf16* mhainw_bf = (bf16*)alloc((size_t)768 * 256 * 2);
  bf16* mhaoutw_bf= (bf16*)alloc((size_t)256 * 256 * 2);

  // activations (with lifetime-based aliasing)
  bf16*  xbf       = (bf16*)alloc((size_t)cB * cS * cH * 2);
  char*  xg_region = (char*)alloc((size_t)cB * cS * 3 * cH * 4);  // 157.3MB
  float* xg        = (float*)xg_region;
  bf16*  qkv_bf    = (bf16*)xg_region;                            // alias: after GRU
  float* hstate    = (float*)alloc((size_t)cB * cH * 4);
  float* hg        = (float*)alloc((size_t)cB * 3 * cH * 4);
  char*  out_region= (char*)alloc((size_t)cB * cS * 2 * cH * 4);  // 104.9MB
  float* outputs   = (float*)out_region;
  bf16*  tfout_bf  = (bf16*)out_region;                           // alias: after LN1
  float* sclin     = (float*)(out_region + (size_t)cB * cS * cD * 2);
  char*  obf_region= (char*)alloc((size_t)cB * cS * 2 * cH * 2);  // 52.4MB
  bf16*  outputs_bf= (bf16*)obf_region;
  bf16*  phon_bf   = (bf16*)obf_region;                           // alias: after QKV gemm
  bf16*  ctx2_bf   = (bf16*)(obf_region + (size_t)cB * cS * cH * 2);
  bf16*  ctx_bf    = (bf16*)alloc((size_t)cB * cS * cD * 2);
  char*  at_region = (char*)alloc((size_t)cB * cS * cD * 4);      // 104.9MB
  float* attnout   = (float*)at_region;
  float* ff2       = (float*)at_region;                           // alias: after LN1
  char*  y_region  = (char*)alloc((size_t)cB * cS * cD * 4);      // 104.9MB
  float* ybuf      = (float*)y_region;
  float* q2        = (float*)y_region;                            // alias: after LN2
  float* ctxout    = (float*)(y_region + (size_t)cB * cS * cH * 4);
  bf16*  y_bf      = (bf16*)alloc((size_t)cB * cS * cD * 2);
  bf16*  ff1_bf    = (bf16*)alloc((size_t)cB * cS * cDFF * 2);
  float* ppout     = (float*)alloc((size_t)cB * cS * cH * 4);
  float* scbuf     = (float*)alloc((size_t)cB * cS * cH * 4);
  bf16*  sc_bf     = (bf16*)alloc((size_t)cB * cS * cH * 2);
  bf16*  enc_bf    = (bf16*)alloc((size_t)cB * cT * cH * 2);
  bf16*  kv_bf     = (bf16*)alloc((size_t)cB * cT * 2 * cH * 2);

  auto cvt = [&](const void* src, bf16* dst, size_t n) {
    k_cvt<<<dim3((unsigned)((n + 255) / 256)), dim3(256), 0, stream>>>((const float*)src, dst, n);
  };

  // weight conversions
  cvt(d_in[4],  wihf_bf,   (size_t)768 * 256);
  cvt(d_in[8],  wihb_bf,   (size_t)768 * 256);
  cvt(d_in[12], tfinw_bf,  (size_t)1536 * 512);
  cvt(d_in[14], tfoutw_bf, (size_t)512 * 512);
  cvt(d_in[16], lin1w_bf,  (size_t)2048 * 512);
  cvt(d_in[18], lin2w_bf,  (size_t)512 * 2048);
  cvt(d_in[24], scw_bf,    (size_t)256 * 512);
  cvt(d_in[28], ppw_bf,    (size_t)256 * 256);
  cvt(d_in[30], mhainw_bf, (size_t)768 * 256);
  cvt(d_in[32], mhaoutw_bf,(size_t)256 * 256);

  // x = syllable + pos
  k_addpos<<<cB * cS * cH / 256, 256, 0, stream>>>(syl, pose, xbf);

  // ---- forward GRU ----
  gemm_bf(stream, xbf, wihf_bf, (const float*)d_in[6], xg, nullptr, cB * cS, 3 * cH, cH, 0);
  k_zero<<<cB * cH / 256, 256, 0, stream>>>(hstate, (size_t)cB * cH);
  for (int s = 0; s < cS; ++s) {
    gemm_f32(stream, hstate, (const float*)d_in[5], (const float*)d_in[7], hg, cB, 3 * cH, cH);
    k_gru_step<<<cB * cH / 256, 256, 0, stream>>>(xg, hg, hstate, outputs, outputs_bf, s, 0);
  }
  // ---- backward GRU ----
  gemm_bf(stream, xbf, wihb_bf, (const float*)d_in[10], xg, nullptr, cB * cS, 3 * cH, cH, 0);
  k_zero<<<cB * cH / 256, 256, 0, stream>>>(hstate, (size_t)cB * cH);
  for (int s = cS - 1; s >= 0; --s) {
    gemm_f32(stream, hstate, (const float*)d_in[9], (const float*)d_in[11], hg, cB, 3 * cH, cH);
    k_gru_step<<<cB * cH / 256, 256, 0, stream>>>(xg, hg, hstate, outputs, outputs_bf, s, cH);
  }

  // ---- transformer encoder layer ----
  gemm_bf(stream, outputs_bf, tfinw_bf, (const float*)d_in[13], nullptr, qkv_bf,
          cB * cS, 3 * cD, cD, 0);
  k_selfattn<<<cB * cNH, 256, 0, stream>>>(qkv_bf, ctx_bf);
  gemm_bf(stream, ctx_bf, tfoutw_bf, (const float*)d_in[15], attnout, nullptr,
          cB * cS, cD, cD, 0);
  k_ln<<<cB * cS, 256, 0, stream>>>(outputs, attnout, (const float*)d_in[20],
                                    (const float*)d_in[21], ybuf, y_bf, nullptr, 0, cD);
  gemm_bf(stream, y_bf, lin1w_bf, (const float*)d_in[17], nullptr, ff1_bf,
          cB * cS, cDFF, cD, 1);
  gemm_bf(stream, ff1_bf, lin2w_bf, (const float*)d_in[19], ff2, nullptr,
          cB * cS, cD, cDFF, 0);
  k_ln<<<cB * cS, 256, 0, stream>>>(ybuf, ff2, (const float*)d_in[22],
                                    (const float*)d_in[23], nullptr, tfout_bf, nullptr, 0, cD);

  // ---- syllable context ----
  gemm_bf(stream, tfout_bf, scw_bf, (const float*)d_in[25], sclin, nullptr,
          cB * cS, cH, cD, 0);
  cvt(phon, phon_bf, (size_t)cB * cS * cH);
  gemm_bf(stream, phon_bf, ppw_bf, (const float*)d_in[29], ppout, nullptr,
          cB * cS, cH, cH, 0);
  k_ln<<<cB * cS, 256, 0, stream>>>(sclin, nullptr, (const float*)d_in[26],
                                    (const float*)d_in[27], scbuf, sc_bf, ppout, 1, cH);

  // ---- cross attention ----
  cvt(enc, enc_bf, (size_t)cB * cT * cH);
  gemm_bf(stream, sc_bf, mhainw_bf, (const float*)d_in[31], q2, nullptr,
          cB * cS, cH, cH, 0);
  gemm_bf(stream, enc_bf, mhainw_bf + (size_t)cH * cH, (const float*)d_in[31] + cH,
          nullptr, kv_bf, cB * cT, 2 * cH, cH, 0);
  k_crossattn<<<cB * cNH * cS, 64, 0, stream>>>(q2, kv_bf, ctx2_bf);
  gemm_bf(stream, ctx2_bf, mhaoutw_bf, (const float*)d_in[33], ctxout, nullptr,
          cB * cS, cH, cH, 0);

  // ---- classifier ----
  k_final<<<cB * cS / 256, 256, 0, stream>>>(scbuf, ctxout, (const float*)d_in[34],
                                             (const float*)d_in[35], (float*)d_out);
}
